// NumpyStyleRNNPolicy_45526653337985
// MI455X (gfx1250) — matrix operand order repaired and checked by reading, not compiled, not run on hardware
//
#include <hip/hip_runtime.h>

#define ALPHA   0.1f
#define B_TOT   1024
#define T_TOT   512
#define OBS_D   64
#define H_D     256
#define A_D     16
#define BT      32        // B rows per workgroup
#define NTHREADS 256      // 8 waves (wave32)

typedef __bf16 bf16;
typedef __bf16 v16bf __attribute__((ext_vector_type(16)));
typedef __bf16 v8bf  __attribute__((ext_vector_type(8)));
typedef float  v8f   __attribute__((ext_vector_type(8)));

__device__ __forceinline__ float sigmoidf(float x) {
    return 1.0f / (1.0f + __expf(-x));
}

__device__ __forceinline__ v16bf frag2(const bf16* p0, const bf16* p1) {
    union { v16bf v; v8bf h[2]; } u;
    u.h[0] = *(const v8bf*)p0;
    u.h[1] = *(const v8bf*)p1;
    return u.v;
}

// A fragment: 16x32 bf16 tile, rows m0.., cols kbase.. of row-major matrix (leading dim ld).
// Lane L: M = L%16, half = L/16; elements 0..7 -> K = kbase+8*half.., 8..15 -> K = kbase+16+8*half..
__device__ __forceinline__ v16bf load_A(const bf16* M, int m0, int ld, int kbase, int lane) {
    const bf16* r = M + (m0 + (lane & 15)) * ld + kbase + ((lane >> 4) << 3);
    return frag2(r, r + 16);
}

// B fragment: B[k][n] = W[n][k] with W row-major [N][ld]. Lane L: N = n0 + L%16,
// elements 0..7 -> K = kbase+8*half.., 8..15 -> K = kbase+16+8*half..
__device__ __forceinline__ v16bf load_B(const bf16* W, int n0, int ld, int kbase, int lane) {
    const bf16* r = W + (n0 + (lane & 15)) * ld + kbase + ((lane >> 4) << 3);
    return frag2(r, r + 16);
}

__global__ __launch_bounds__(NTHREADS, 1)
void rnn_policy_scan_kernel(const float* __restrict__ obs,
                            const float* __restrict__ gWin,
                            const float* __restrict__ gWh,
                            const float* __restrict__ gbh,
                            const float* __restrict__ gWout,
                            const float* __restrict__ gbo,
                            float* __restrict__ out) {
    extern __shared__ char smem[];
    bf16*  sWh  = (bf16*)smem;                 // [H][H]     128 KB
    bf16*  sWin = sWh  + H_D * H_D;            // [H][OBS]    32 KB
    bf16*  sWo  = sWin + H_D * OBS_D;          // [A][H]       8 KB
    bf16*  sH   = sWo  + A_D * H_D;            // [BT][H]     16 KB
    bf16*  sX   = sH   + BT  * H_D;            // [BT][OBS]    4 KB
    float* sbh  = (float*)(sX + BT * OBS_D);   // [H]          1 KB
    float* sbo  = sbh + H_D;                   // [A]

    const int tid  = threadIdx.x;
    const int lane = tid & 31;
    const int w    = tid >> 5;       // wave id 0..7
    const int hs   = lane >> 4;      // lane-half select
    const int ln   = lane & 15;      // N index within tile
    const int b0   = blockIdx.x * BT;

    // ---- one-time: weights -> LDS (bf16), biases -> LDS (f32), h0 = 0 ----
    for (int i = tid; i < H_D * H_D;   i += NTHREADS) sWh[i]  = (bf16)gWh[i];
    for (int i = tid; i < H_D * OBS_D; i += NTHREADS) sWin[i] = (bf16)gWin[i];
    for (int i = tid; i < A_D * H_D;   i += NTHREADS) sWo[i]  = (bf16)gWout[i];
    for (int i = tid; i < H_D;         i += NTHREADS) sbh[i]  = gbh[i];
    for (int i = tid; i < A_D;         i += NTHREADS) sbo[i]  = gbo[i];
    for (int i = tid; i < BT * H_D;    i += NTHREADS) sH[i]   = (bf16)0.0f;
    __syncthreads();

    // tile assignment: pre_h is [BT=32][H=256] = 2 x 16 WMMA tiles, 4 per wave
    const int mtile = w & 1;
    const int nt0   = (w >> 1) << 2;

    float* h_out = out + (size_t)B_TOT * T_TOT * A_D;
    float* o_out = h_out + (size_t)B_TOT * H_D;

    v8f o_acc = {0.f, 0.f, 0.f, 0.f, 0.f, 0.f, 0.f, 0.f};  // persistent o state (waves 0,1)

    // obs staging: 8 threads per B-row, 8 contiguous floats each
    const int xrow = tid >> 3;
    const int xcol = (tid & 7) << 3;
    const float* xsrc = obs + (size_t)(b0 + xrow) * T_TOT * OBS_D + xcol;

    for (int t = 0; t < T_TOT; ++t) {
        // ---- stage obs[:, t, :] tile into LDS (f32 -> bf16) ----
        {
            const float4* p = (const float4*)(xsrc + (size_t)t * OBS_D);
            float4 a = p[0], b = p[1];
            v8bf xv;
            xv[0] = (bf16)a.x; xv[1] = (bf16)a.y; xv[2] = (bf16)a.z; xv[3] = (bf16)a.w;
            xv[4] = (bf16)b.x; xv[5] = (bf16)b.y; xv[6] = (bf16)b.z; xv[7] = (bf16)b.w;
            *(v8bf*)(sX + xrow * OBS_D + xcol) = xv;
            if (t + 1 < T_TOT)
                __builtin_prefetch(xsrc + (size_t)(t + 1) * OBS_D, 0, 1);  // global_prefetch_b8
        }
        __syncthreads();

        // ---- pre = b_h + obs_t @ W_in^T + h @ W_h^T  (this wave's 16x64 tile) ----
        v8f acc[4];
        #pragma unroll
        for (int j = 0; j < 4; ++j) {
            float bv = sbh[(nt0 + j) * 16 + ln];
            #pragma unroll
            for (int r = 0; r < 8; ++r) acc[j][r] = bv;
        }
        #pragma unroll
        for (int kk = 0; kk < 2; ++kk) {                       // K over OBS (64)
            v16bf af = load_A(sX, mtile * 16, OBS_D, kk * 32, lane);
            #pragma unroll
            for (int j = 0; j < 4; ++j) {
                v16bf bfv = load_B(sWin, (nt0 + j) * 16, OBS_D, kk * 32, lane);
                acc[j] = __builtin_amdgcn_wmma_f32_16x16x32_bf16(
                    false, af, false, bfv, (short)0, acc[j], false, false);
            }
        }
        #pragma unroll
        for (int kk = 0; kk < 8; ++kk) {                       // K over H (256)
            v16bf af = load_A(sH, mtile * 16, H_D, kk * 32, lane);
            #pragma unroll
            for (int j = 0; j < 4; ++j) {
                v16bf bfv = load_B(sWh, (nt0 + j) * 16, H_D, kk * 32, lane);
                acc[j] = __builtin_amdgcn_wmma_f32_16x16x32_bf16(
                    false, af, false, bfv, (short)0, acc[j], false, false);
            }
        }
        __syncthreads();   // all waves finished reading sH as A fragments

        // ---- h = (1-a)*h + a*sigmoid(pre)  (each wave's region is exclusive) ----
        #pragma unroll
        for (int j = 0; j < 4; ++j) {
            #pragma unroll
            for (int r = 0; r < 8; ++r) {
                int M = mtile * 16 + r + 8 * hs;
                int N = (nt0 + j) * 16 + ln;
                float hold = (float)sH[M * H_D + N];
                float hn = (1.0f - ALPHA) * hold + ALPHA * sigmoidf(acc[j][r]);
                sH[M * H_D + N] = (bf16)hn;
            }
        }
        __syncthreads();   // new h visible to everyone

        // ---- waves 0,1: pre_o = h @ W_out^T + b_out ; o update ; stream out ----
        if (w < 2) {
            float bo = sbo[ln];
            v8f oc;
            #pragma unroll
            for (int r = 0; r < 8; ++r) oc[r] = bo;
            #pragma unroll
            for (int kk = 0; kk < 8; ++kk) {
                v16bf af  = load_A(sH, w * 16, H_D, kk * 32, lane);
                v16bf bfv = load_B(sWo, 0, H_D, kk * 32, lane);
                oc = __builtin_amdgcn_wmma_f32_16x16x32_bf16(
                    false, af, false, bfv, (short)0, oc, false, false);
            }
            #pragma unroll
            for (int r = 0; r < 8; ++r) {
                float on = (1.0f - ALPHA) * o_acc[r] + ALPHA * sigmoidf(oc[r]);
                o_acc[r] = on;
                int bb = b0 + w * 16 + r + 8 * hs;
                out[((size_t)bb * T_TOT + t) * A_D + ln] = on;
            }
        }
    }

    // ---- final states: h [1,B,H] then o [1,B,A] appended after out ----
    __syncthreads();
    for (int i = tid; i < BT * H_D; i += NTHREADS) {
        int row = i / H_D, col = i % H_D;
        h_out[(size_t)(b0 + row) * H_D + col] = (float)sH[i];
    }
    if (w < 2) {
        #pragma unroll
        for (int r = 0; r < 8; ++r) {
            int bb = b0 + w * 16 + r + 8 * hs;
            o_out[(size_t)bb * A_D + ln] = o_acc[r];
        }
    }
}

extern "C" void kernel_launch(void* const* d_in, const int* in_sizes, int n_in,
                              void* d_out, int out_size, void* d_ws, size_t ws_size,
                              hipStream_t stream) {
    (void)in_sizes; (void)n_in; (void)out_size; (void)d_ws; (void)ws_size;
    const float* obs  = (const float*)d_in[0];
    const float* Win  = (const float*)d_in[1];
    const float* Wh   = (const float*)d_in[2];
    const float* bh   = (const float*)d_in[3];
    const float* Wout = (const float*)d_in[4];
    const float* bout = (const float*)d_in[5];
    float* out = (float*)d_out;

    size_t smem = (size_t)(H_D * H_D + H_D * OBS_D + A_D * H_D + BT * H_D + BT * OBS_D) * sizeof(bf16)
                + (size_t)(H_D + A_D) * sizeof(float);   // ~189.6 KB, fits 320 KB WGP LDS

    hipFuncSetAttribute(reinterpret_cast<const void*>(rnn_policy_scan_kernel),
                        hipFuncAttributeMaxDynamicSharedMemorySize, (int)smem);

    rnn_policy_scan_kernel<<<B_TOT / BT, NTHREADS, smem, stream>>>(
        obs, Win, Wh, bh, Wout, bout, out);
}
